// LSTM_MLP_34342558498963
// MI455X (gfx1250) — matrix operand order repaired and checked by reading, not compiled, not run on hardware
//
#include <hip/hip_runtime.h>
#include <hip/hip_bf16.h>

// ---------------------------------------------------------------------------
// LSTM (B=256, D=256, T=512, H=512) + per-step Linear(H,1)+sigmoid, for
// MI455X (gfx1250). Single persistent kernel runs all 512 recurrent steps
// with one device-wide barrier per step; bf16 WMMA (v_wmma_f32_16x16x32_bf16)
// with f32 accumulation; W_hh/W_ih tiles resident in LDS (152 KB/WG of the
// 320 KB WGP LDS); cell state c held in registers for the whole sequence.
// Per-step h/x staging uses GLOBAL_LOAD_ASYNC_TO_LDS_B128 (ASYNCcnt) when the
// builtin is available, plus global_prefetch of the next timestep's x slice.
// ---------------------------------------------------------------------------

#define Bsz 256
#define Dd  256
#define Tt  512
#define Hh  512
#define G4H 2048
#define NWG 256   // 8 batch-tiles x 32 hidden-tiles, persistent

#if defined(__AMDGCN__) && __has_builtin(__builtin_amdgcn_global_load_async_to_lds_b128) && \
    __has_builtin(__builtin_amdgcn_s_wait_asynccnt)
#define ASYNC_LDS 1
#else
#define ASYNC_LDS 0
#endif

typedef __attribute__((ext_vector_type(16))) __bf16 v16bf;
typedef __attribute__((ext_vector_type(8)))  float  v8f;
typedef int v4i __attribute__((__vector_size__(16)));
#if ASYNC_LDS
typedef __attribute__((address_space(1))) v4i* as1_v4i;
typedef __attribute__((address_space(3))) v4i* as3_v4i;
#endif

union F32B { struct { uint4 lo, hi; } q; v16bf v; };
union F8U  { uint4 q[2]; v8f v; };

__device__ __forceinline__ unsigned short f2bf(float f) {
  unsigned u = __builtin_bit_cast(unsigned, f);
  u += 0x7FFFu + ((u >> 16) & 1u);          // round-to-nearest-even
  return (unsigned short)(u >> 16);
}
__device__ __forceinline__ float sigm(float x) { return 1.f / (1.f + __expf(-x)); }
__device__ __forceinline__ float tanh_(float x) {
  float y = fminf(fmaxf(x, -12.f), 12.f);
  float e = __expf(2.f * y);
  return (e - 1.f) / (e + 1.f);
}

// 16B memory -> LDS copy: async (direct-to-LDS, ASYNCcnt) when available,
// else sync load + ds_store fallback.
// Generic->AS3 via low-32-bit truncation is valid on CDNA5: the LDS aperture
// uses the low 32 bits as the LDS address; generic->AS1 is an identity reinterpret.
__device__ __forceinline__ void cp16_g2l(void* l, const void* g) {
#if ASYNC_LDS
  __builtin_amdgcn_global_load_async_to_lds_b128(
      (as1_v4i)(unsigned long long)g,
      (as3_v4i)(unsigned int)(unsigned long long)l, 0, 0);
#else
  *(uint4*)l = *(const uint4*)g;
#endif
}
__device__ __forceinline__ void cp_wait() {
#if ASYNC_LDS
  __builtin_amdgcn_s_wait_asynccnt(0);
#endif
}

// A-fragment (16x32 bf16): lanes 0-15 rows M=0..15 with K {kb..kb+7, kb+16..kb+23};
// lanes 16-31 same rows, K shifted by 8.  Two 16B LDS loads per lane.
__device__ __forceinline__ v16bf ldA(const unsigned short* base, int row, int stride,
                                     int kb, bool hi) {
  F32B u;
  const unsigned short* p = base + row * stride + kb + (hi ? 8 : 0);
  u.q.lo = *(const uint4*)p;
  u.q.hi = *(const uint4*)(p + 16);
  return u.v;
}
// B-fragment (32x16 bf16): lane n holds column n; lanes 0-15 K {kb..kb+7, kb+16..kb+23},
// lanes 16-31 K shifted by 8 (same order as A).
__device__ __forceinline__ v16bf ldB(const unsigned short* base, int crow, int stride,
                                     int kb, bool hi) {
  F32B u;
  const unsigned short* p = base + crow * stride + kb + (hi ? 8 : 0);
  u.q.lo = *(const uint4*)p;
  u.q.hi = *(const uint4*)(p + 16);
  return u.v;
}
__device__ __forceinline__ v8f ldP(const float* sp, int wv, int lane) {
  F8U u;
  const uint4* p = (const uint4*)&sp[wv * 256 + lane * 8];
  u.q[0] = p[0]; u.q[1] = p[1];
  return u.v;
}

// --------------------------- helper kernels --------------------------------

__global__ void k_init(unsigned int* p, int n) {
  int i = blockIdx.x * 256 + threadIdx.x;
  if (i < n) p[i] = 0u;
}

__global__ void k_castw(const float* __restrict__ wih, const float* __restrict__ whh,
                        const float* __restrict__ bih, const float* __restrict__ bhh,
                        unsigned short* __restrict__ wihb, unsigned short* __restrict__ whhb,
                        float* __restrict__ bias) {
  int i = blockIdx.x * 256 + threadIdx.x;
  if (i < G4H * Dd) wihb[i] = f2bf(wih[i]);
  if (i < G4H * Hh) whhb[i] = f2bf(whh[i]);
  if (i < G4H)      bias[i] = bih[i] + bhh[i];
}

// x (B,D,T) f32 -> xt (T,B,D) bf16, 32x32 LDS tile transpose (coalesced both sides)
__global__ void k_xT(const float* __restrict__ x, unsigned short* __restrict__ xt) {
  __shared__ float tile[32][33];
  int b = blockIdx.z, d0 = blockIdx.y * 32, t0 = blockIdx.x * 32;
  int tx = threadIdx.x, ty = threadIdx.y;     // 32 x 8
  const float* src = x + ((size_t)b * Dd + d0) * Tt + t0;
#pragma unroll
  for (int k = 0; k < 4; ++k)
    tile[ty + 8 * k][tx] = src[(size_t)(ty + 8 * k) * Tt + tx];
  __syncthreads();
#pragma unroll
  for (int k = 0; k < 4; ++k) {
    int tr = ty + 8 * k;
    xt[((size_t)(t0 + tr) * Bsz + b) * Dd + d0 + tx] = f2bf(tile[tx][tr]);
  }
}

__global__ void k_out(const float* __restrict__ logits, const float* __restrict__ bmlp,
                      float* __restrict__ out) {
  int i = blockIdx.x * 256 + threadIdx.x;
  if (i < Bsz * Tt) {
    int b = i / Tt, t = i % Tt;
    out[i] = 1.f / (1.f + __expf(-(logits[t * Bsz + b] + bmlp[0])));
  }
}

// --------------------------- persistent LSTM -------------------------------
// Grid: 256 WGs = 8 (batch tiles of 32) x 32 (hidden tiles of 16), 256 thr/WG.
// Wave w: mtile = w&1 (16 batch rows), gate = w>>1 (i,f,g,o). Each wave owns a
// 16x16 output fragment, K-loop = 256 (x part) + 512 (h part) => 24 WMMA/step.
__global__ __launch_bounds__(256, 1)
void k_lstm(const unsigned short* __restrict__ xt, const unsigned short* __restrict__ wihb,
            const unsigned short* __restrict__ whhb, const float* __restrict__ bias,
            const float* __restrict__ wmlp, unsigned short* __restrict__ hbuf,
            float* __restrict__ logits, int* __restrict__ bar) {
  __shared__ unsigned short s_whh[64 * 512];   // 64 KB : 4 gates x 16 cols, K=512
  __shared__ unsigned short s_wih[64 * 256];   // 32 KB : 4 gates x 16 cols, K=256
  __shared__ unsigned short s_h[32 * 512];     // 32 KB : h tile, per step
  __shared__ unsigned short s_x[32 * 256];     // 16 KB : x slice, per step
  __shared__ float          s_p[8 * 256];      //  8 KB : gate pre-activations

  const int wg = blockIdx.x;
  const int mb = wg >> 5, jt = wg & 31;
  const int b0 = mb * 32, j0 = jt * 16;
  const int tid = threadIdx.x;
  const int w = tid >> 5, lane = tid & 31;
  const int mtile = w & 1, gate = w >> 1;
  const int l15 = lane & 15;
  const bool hi = lane >= 16;

  // ---- stage weight tiles into LDS once (resident for all 512 steps) ----
  {
    for (int i = tid; i < 4096; i += 256) {           // 64 rows x 64 uint4
      int c = i >> 6, q = i & 63;
      int n = (c >> 4) * Hh + j0 + (c & 15);
      cp16_g2l((uint4*)s_whh + i, whhb + (size_t)n * Hh + q * 8);
    }
    for (int i = tid; i < 2048; i += 256) {           // 64 rows x 32 uint4
      int c = i >> 5, q = i & 31;
      int n = (c >> 4) * Hh + j0 + (c & 15);
      cp16_g2l((uint4*)s_wih + i, wihb + (size_t)n * Dd + q * 8);
    }
  }

  const float wm = wmlp[j0 + l15];
  const float bv = bias[gate * Hh + j0 + l15];
  const int arow = mtile * 16 + l15;     // A row for this lane
  const int crow = gate * 16 + l15;      // B (weight) LDS row for this lane

  v8f cst;                               // persistent cell state (combiner waves)
#pragma unroll
  for (int i = 0; i < 8; ++i) cst[i] = 0.f;

  for (int t = 0; t < Tt; ++t) {
    const unsigned short* hcur = hbuf + (size_t)(t & 1) * (Bsz * Hh);
    unsigned short* hnx = hbuf + (size_t)((t + 1) & 1) * (Bsz * Hh);

    // ---- stage h tile (32x512) and x slice (32x256) into LDS (async) ----
    {
      for (int i = tid; i < 2048; i += 256) {          // 32 rows x 64 uint4
        int r = i >> 6, q = i & 63;
        cp16_g2l((uint4*)s_h + i, hcur + (size_t)(b0 + r) * Hh + q * 8);
      }
      const unsigned short* xsrc = xt + ((size_t)t * Bsz + b0) * Dd;
      for (int i = tid; i < 1024; i += 256) {          // 32 rows x 32 uint4
        int r = i >> 5, q = i & 31;
        cp16_g2l((uint4*)s_x + i, xsrc + (size_t)r * Dd + q * 8);
      }
      // prefetch next step's x slice into cache (barrier-independent work)
      if (t + 1 < Tt) {
        const unsigned short* xnxt = xt + ((size_t)(t + 1) * Bsz + b0) * Dd;
        __builtin_prefetch((const char*)xnxt + tid * 64, 0, 1);
      }
    }
    cp_wait();
    __syncthreads();

    // ---- gate pre-activation GEMM: acc = bias + x@Wih^T + h@Whh^T ----
    v8f acc;
#pragma unroll
    for (int i = 0; i < 8; ++i) acc[i] = bv;
#pragma unroll
    for (int kb = 0; kb < Dd; kb += 32) {
      v16bf a = ldA(s_x, arow, Dd, kb, hi);
      v16bf b = ldB(s_wih, crow, Dd, kb, hi);
      acc = __builtin_amdgcn_wmma_f32_16x16x32_bf16(false, a, false, b,
                                                    (short)0, acc, false, false);
    }
#pragma unroll
    for (int kb = 0; kb < Hh; kb += 32) {
      v16bf a = ldA(s_h, arow, Hh, kb, hi);
      v16bf b = ldB(s_whh, crow, Hh, kb, hi);
      acc = __builtin_amdgcn_wmma_f32_16x16x32_bf16(false, a, false, b,
                                                    (short)0, acc, false, false);
    }

    // ---- share pre-activations across gate waves ----
    {
      F8U u; u.v = acc;
      uint4* d = (uint4*)&s_p[w * 256 + lane * 8];
      d[0] = u.q[0]; d[1] = u.q[1];
    }
    __syncthreads();

    // ---- combiner waves (gate 0): elementwise LSTM cell + output dot ----
    if (w < 2) {
      v8f Pi = acc;                       // own fragment is gate i
      v8f Pf = ldP(s_p, 2 + w, lane);
      v8f Pg = ldP(s_p, 4 + w, lane);
      v8f Po = ldP(s_p, 6 + w, lane);
#pragma unroll
      for (int r = 0; r < 8; ++r) {
        float iv = sigm(Pi[r]);
        float fv = sigm(Pf[r]);
        float gv = tanh_(Pg[r]);
        float ov = sigm(Po[r]);
        float cn = fv * cst[r] + iv * gv;
        cst[r] = cn;
        float hv = ov * tanh_(cn);
        int brow = b0 + w * 16 + r + (hi ? 8 : 0);
        hnx[(size_t)brow * Hh + j0 + l15] = f2bf(hv);
        // logits partial: reduce over the 16 hidden cols held across lanes
        float pp = hv * wm;
        pp += __shfl_xor(pp, 1);
        pp += __shfl_xor(pp, 2);
        pp += __shfl_xor(pp, 4);
        pp += __shfl_xor(pp, 8);
        if (l15 == 0) atomicAdd(&logits[t * Bsz + brow], pp);
      }
    }

    // ---- device-wide barrier: one per timestep ----
    __threadfence();
    __syncthreads();
    if (tid == 0) {
      atomicAdd(bar + t, 1);
      while (__hip_atomic_load(bar + t, __ATOMIC_ACQUIRE, __HIP_MEMORY_SCOPE_AGENT) < NWG)
        __builtin_amdgcn_s_sleep(2);
    }
    __syncthreads();
  }
}

// ------------------------------ launcher -----------------------------------

extern "C" void kernel_launch(void* const* d_in, const int* in_sizes, int n_in,
                              void* d_out, int out_size, void* d_ws, size_t ws_size,
                              hipStream_t stream) {
  const float* x    = (const float*)d_in[0];
  const float* Wih  = (const float*)d_in[1];
  const float* Whh  = (const float*)d_in[2];
  const float* bih  = (const float*)d_in[3];
  const float* bhh  = (const float*)d_in[4];
  const float* Wmlp = (const float*)d_in[5];
  const float* bmlp = (const float*)d_in[6];
  float* out = (float*)d_out;

  char* ws = (char*)d_ws;
  size_t off = 0;
  auto alloc = [&](size_t bytes) {
    char* p = ws + off;
    off += (bytes + 255) & ~(size_t)255;
    return p;
  };
  unsigned short* xt   = (unsigned short*)alloc((size_t)Tt * Bsz * Dd * 2);  // 64 MB
  unsigned short* wihb = (unsigned short*)alloc((size_t)G4H * Dd * 2);       // 1 MB
  unsigned short* whhb = (unsigned short*)alloc((size_t)G4H * Hh * 2);       // 2 MB
  float*          bias = (float*)alloc((size_t)G4H * 4);
  // zero region starts here: h double-buffer, logits, barrier counters
  unsigned short* hbuf   = (unsigned short*)alloc((size_t)2 * Bsz * Hh * 2);
  float*          logits = (float*)alloc((size_t)Tt * Bsz * 4);
  int*            bar    = (int*)alloc((size_t)Tt * 4);

  // zero h/logits/bar every launch (graph replays need deterministic state)
  int nz = (2 * Bsz * Hh * 2 + Tt * Bsz * 4 + Tt * 4) / 4;
  k_init<<<(nz + 255) / 256, 256, 0, stream>>>((unsigned int*)hbuf, nz);

  k_castw<<<(G4H * Hh + 255) / 256, 256, 0, stream>>>(Wih, Whh, bih, bhh,
                                                      wihb, whhb, bias);
  k_xT<<<dim3(Tt / 32, Dd / 32, Bsz), dim3(32, 8), 0, stream>>>(x, xt);

  k_lstm<<<NWG, 256, 0, stream>>>(xt, wihb, whhb, bias, Wmlp, hbuf, logits, bar);

  k_out<<<(Bsz * Tt + 255) / 256, 256, 0, stream>>>(logits, bmlp, out);
}
